// OmelyanIntegrator_72593537237652
// MI455X (gfx1250) — matrix operand order repaired and checked by reading, not compiled, not run on hardware
//
#include <hip/hip_runtime.h>
#include <hip/hip_bf16.h>
#include <math.h>

typedef __attribute__((ext_vector_type(16))) _Float16 v16h;
typedef __attribute__((ext_vector_type(8)))  _Float16 v8h;
typedef __attribute__((ext_vector_type(8)))  float    v8f;

#define DIM   1024
#define RANK  256
#define MT    16          // batch rows per workgroup (one WMMA M-tile)
#define KT_U  (DIM / 32)  // 32 K-tiles for GEMM1
#define NT_U  (RANK / 16) // 16 N-tiles for GEMM1
#define KT_W  (RANK / 32) // 8  K-tiles for GEMM2
#define NT_W  (DIM / 16)  // 64 N-tiles for GEMM2
#define LANE_H 24         // halfs per lane slot in A-staging (48B: 16B-aligned, bank-skewed)
#define TILE_H (32 * LANE_H)

// ---- WMMA A-operand swizzle (CDNA5 ISA 16-bit A layout) ----
__device__ __forceinline__ int a_swz_idx(int kt, int M, int K_local) {
    int half = (K_local >> 3) & 1;
    int K0   = K_local - 8 * half;
    int e    = (K0 < 8) ? K0 : K0 - 8;
    return (kt * 32 + (M + 16 * half)) * LANE_H + e;
}

__device__ __forceinline__ v16h load_a_swz(const _Float16* stg, int kt, int lane) {
    const _Float16* p = stg + (kt * 32 + lane) * LANE_H;
    v8h lo = *(const v8h*)p;
    v8h hi = *(const v8h*)(p + 8);
    return __builtin_shufflevector(lo, hi, 0, 1, 2, 3, 4, 5, 6, 7,
                                   8, 9, 10, 11, 12, 13, 14, 15);
}

// ---- Prologue: convert f32 U/W ([K][N] row-major) into WMMA-B f16 tiles ----
// B layout: lane -> N = lane&15 ; K = (elem<8 ? elem : elem+8) + 8*(lane>>4)
__global__ void omelyan_cvt_b(const float* __restrict__ src, _Float16* __restrict__ dst,
                              int K, int N, int KT) {
    int tid = blockIdx.x * blockDim.x + threadIdx.x;
    int total = (N / 16) * KT * 32;
    if (tid >= total) return;
    int lane = tid & 31;
    int tile = tid >> 5;                 // tile = nt*KT + kt
    int kt   = tile % KT;
    int half = lane >> 4;
    int nt   = tile / KT;
    int n    = nt * 16 + (lane & 15);
    _Float16* out = dst + (size_t)tile * 512 + lane * 16;
#pragma unroll
    for (int e = 0; e < 16; ++e) {
        int k = kt * 32 + (e < 8 ? e : e + 8) + 8 * half;
        out[e] = (_Float16)src[(size_t)k * N + n];
    }
}

// ---- Main integrator: 16 rows per WG; x,v in registers (C/D layout) ----
__global__ void __launch_bounds__(256)
omelyan_kernel(const float* __restrict__ x_in, const float* __restrict__ v_in,
               const float* __restrict__ force, const float* __restrict__ Vw,
               const _Float16* __restrict__ Uh, const _Float16* __restrict__ Wh,
               const int* __restrict__ steps_ptr,
               float* __restrict__ out_x, float* __restrict__ out_v) {
    __shared__ _Float16 vstg[KT_U * TILE_H];   // v (f16), A-swizzled, 48 KB
    __shared__ _Float16 h2stg[KT_W * TILE_H];  // h^2 (f16), A-swizzled, 12 KB
    __shared__ float    Vws[DIM];              // 4 KB
    __shared__ float    r2s[MT];

    const int t    = threadIdx.x;
    const int wave = t >> 5;
    const int lane = t & 31;
    const int half = lane >> 4;
    const int ln   = lane & 15;
    const int m0   = blockIdx.x * MT;

    // element (j,r) lives at row M = r + 8*half, col d = (wave*8+j)*16 + ln
    float xr[8][8], vr[8][8];

#pragma unroll
    for (int j = 0; j < 8; ++j) {
        const int d  = (wave * 8 + j) * 16 + ln;
        const int kt = d >> 5, Kl = d & 31;
#pragma unroll
        for (int r = 0; r < 8; ++r) {
            const int M = r + 8 * half;
            xr[j][r] = x_in[(size_t)(m0 + M) * DIM + d];
            vr[j][r] = v_in[(size_t)(m0 + M) * DIM + d];
            vstg[a_swz_idx(kt, M, Kl)] = (_Float16)vr[j][r];
        }
    }
    for (int j = t; j < DIM; j += 256) Vws[j] = Vw[j];
    __syncthreads();

    const int   steps = *steps_ptr;
    const float dt    = 0.01f;
    const float Cc[4] = {0.1786178958448091f, -0.0662645826698185f,
                         0.7752933736500188f, -0.0662645826698185f};
    const float Dd[4] = {0.7123418310626054f, -0.2123418310626054f,
                         -0.2123418310626054f, 0.7123418310626054f};
    const float C5 = 0.1786178958448091f;

    const int nt0 = wave * 2, nt1 = wave * 2 + 1;

    for (int s = 0; s < steps; ++s) {
#pragma unroll 1
        for (int sub = 0; sub < 4; ++sub) {
            const float cdt = Cc[sub] * dt;
            const float ddt = Dd[sub] * dt;

            if (t < MT) r2s[t] = 0.0f;
            __syncthreads();

            // ---- x += c*dt*v (registers) ; r2 row sums via LDS atomics ----
            {
                float part[8];
#pragma unroll
                for (int r = 0; r < 8; ++r) part[r] = 0.0f;
#pragma unroll
                for (int j = 0; j < 8; ++j)
#pragma unroll
                    for (int r = 0; r < 8; ++r) {
                        float nx = xr[j][r] + cdt * vr[j][r];
                        xr[j][r] = nx;
                        part[r] += nx * nx;
                    }
#pragma unroll
                for (int r = 0; r < 8; ++r)
                    atomicAdd(&r2s[r + 8 * half], part[r]);
            }

            // ---- GEMM1: h = v_f16 @ Uh ; stage h^2 (A-swizzled) ----
            {
                v8f acc0 = {}, acc1 = {};
                const v16h* pB0 = (const v16h*)Uh + (size_t)(nt0 * KT_U) * 32 + lane;
                const v16h* pB1 = (const v16h*)Uh + (size_t)(nt1 * KT_U) * 32 + lane;
#pragma unroll 4
                for (int kt = 0; kt < KT_U; ++kt) {
                    v16h A  = load_a_swz(vstg, kt, lane);
                    v16h B0 = pB0[kt * 32];
                    v16h B1 = pB1[kt * 32];
                    acc0 = __builtin_amdgcn_wmma_f32_16x16x32_f16(false, A, false, B0,
                                                                  (short)0, acc0, false, false);
                    acc1 = __builtin_amdgcn_wmma_f32_16x16x32_f16(false, A, false, B1,
                                                                  (short)0, acc1, false, false);
                }
                const int n0 = nt0 * 16 + ln, n1 = nt1 * 16 + ln;
                const int kt0 = n0 >> 5, Kl0 = n0 & 31;
                const int kt1 = n1 >> 5, Kl1 = n1 & 31;
#pragma unroll
                for (int r = 0; r < 8; ++r) {
                    const int M = r + 8 * half;
                    float h0 = acc0[r], h1 = acc1[r];
                    h2stg[a_swz_idx(kt0, M, Kl0)] = (_Float16)(h0 * h0);
                    h2stg[a_swz_idx(kt1, M, Kl1)] = (_Float16)(h1 * h1);
                }
            }
            __syncthreads();

            // ---- GEMM2 in 2 passes of 4 d-tiles (keeps VGPRs < 256) ----
            float sing[8];
#pragma unroll
            for (int r = 0; r < 8; ++r)
                sing[r] = 1.0f + __expf(-r2s[r + 8 * half]);

#pragma unroll 1
            for (int pass = 0; pass < 2; ++pass) {
                v8f acc[4];
#pragma unroll
                for (int jj = 0; jj < 4; ++jj) acc[jj] = (v8f){};
#pragma unroll
                for (int kt = 0; kt < KT_W; ++kt) {
                    v16h A = load_a_swz(h2stg, kt, lane);
                    v16h Bt[4];
#pragma unroll
                    for (int jj = 0; jj < 4; ++jj) {
                        int dtile = wave * 8 + pass * 4 + jj;
                        Bt[jj] = ((const v16h*)Wh)[(size_t)(dtile * KT_W + kt) * 32 + lane];
                    }
#pragma unroll
                    for (int jj = 0; jj < 4; ++jj)
                        acc[jj] = __builtin_amdgcn_wmma_f32_16x16x32_f16(false, A, false, Bt[jj],
                                                                         (short)0, acc[jj], false, false);
                }
#pragma unroll
                for (int jj = 0; jj < 4; ++jj) {
                    const int   j   = pass * 4 + jj;
                    const int   d   = (wave * 8 + j) * 16 + ln;
                    const int   kt  = d >> 5, Kl = d & 31;
                    const float Vwd = Vws[d];
#pragma unroll
                    for (int r = 0; r < 8; ++r) {
                        const int M = r + 8 * half;
                        float f    = force[(size_t)(m0 + M) * DIM + d];
                        // gate = 1 + 0.1*tanh(z) = 1.1 - 0.2/(1+e^{2z})
                        float e2z  = __expf(2.0f * xr[j][r] * Vwd);
                        float gate = 1.1f - 0.2f * __builtin_amdgcn_rcpf(1.0f + e2z);
                        float a    = f - acc[jj][r] * gate * sing[r];
                        vr[j][r] += ddt * a;
                        vstg[a_swz_idx(kt, M, Kl)] = (_Float16)vr[j][r];
                    }
                }
            }
            __syncthreads();
        }
        // ---- trailing x += C5*dt*v (pure registers) ----
#pragma unroll
        for (int j = 0; j < 8; ++j)
#pragma unroll
            for (int r = 0; r < 8; ++r)
                xr[j][r] += C5 * dt * vr[j][r];
    }

    // ---- store results ----
#pragma unroll
    for (int j = 0; j < 8; ++j) {
        const int d = (wave * 8 + j) * 16 + ln;
#pragma unroll
        for (int r = 0; r < 8; ++r) {
            const int M = r + 8 * half;
            out_x[(size_t)(m0 + M) * DIM + d] = xr[j][r];
            out_v[(size_t)(m0 + M) * DIM + d] = vr[j][r];
        }
    }
}

extern "C" void kernel_launch(void* const* d_in, const int* in_sizes, int n_in,
                              void* d_out, int out_size, void* d_ws, size_t ws_size,
                              hipStream_t stream) {
    const float* x     = (const float*)d_in[0];
    const float* v     = (const float*)d_in[1];
    const float* force = (const float*)d_in[2];
    const float* U     = (const float*)d_in[3];  // [DIM][RANK]
    const float* W     = (const float*)d_in[4];  // [RANK][DIM]
    const float* Vw    = (const float*)d_in[5];  // [DIM]
    const int*   steps = (const int*)d_in[6];

    const int B = in_sizes[0] / DIM;             // 4096

    // workspace: f16 WMMA-B-swizzled U then W (512 KB each)
    _Float16* Uh = (_Float16*)d_ws;
    _Float16* Wh = Uh + (size_t)DIM * RANK;

    {
        int totalU = NT_U * KT_U * 32;           // 16384 threads
        int totalW = NT_W * KT_W * 32;           // 16384 threads
        omelyan_cvt_b<<<(totalU + 255) / 256, 256, 0, stream>>>(U, Uh, DIM, RANK, KT_U);
        omelyan_cvt_b<<<(totalW + 255) / 256, 256, 0, stream>>>(W, Wh, RANK, DIM, KT_W);
    }

    float* out_x = (float*)d_out;
    float* out_v = out_x + (size_t)B * DIM;

    omelyan_kernel<<<B / MT, 256, 0, stream>>>(x, v, force, Vw, Uh, Wh, steps, out_x, out_v);
}
